// GravNetLayer_14267881357574
// MI455X (gfx1250) — matrix operand order repaired and checked by reading, not compiled, not run on hardware
//
#include <hip/hip_runtime.h>

#define BATCH 8
#define BN    2048
#define NFEAT 64
#define KNN   16
#define HID   32

typedef __attribute__((ext_vector_type(16))) _Float16 v16h;
typedef __attribute__((ext_vector_type(8)))  _Float16 v8h;
typedef __attribute__((ext_vector_type(2)))  _Float16 v2h;
typedef __attribute__((ext_vector_type(8)))  float    v8f;

// -------------------------------------------------------------------------
// Kernel 1: brute-force kNN (K=16) on 2-D positions, per batch.
// One thread per query point; candidate positions tiled through LDS.
// Top-K kept in registers via replace-max + unrolled select (no scratch).
// -------------------------------------------------------------------------
__global__ __launch_bounds__(256) void knn_kernel(const float* __restrict__ x,
                                                  int* __restrict__ knn) {
    __shared__ float spos[256][2];
    const int b = blockIdx.x >> 3;                       // 8 chunks of 256 per batch
    const int i = ((blockIdx.x & 7) << 8) + threadIdx.x; // query point in batch
    const int rowq = (b * BN + i) * 66;
    const float qx = x[rowq + 0];
    const float qy = x[rowq + 1];

    float bd[KNN];
    int   bi[KNN];
#pragma unroll
    for (int t = 0; t < KNN; ++t) { bd[t] = 3.0e38f; bi[t] = 0; }
    float bmax = 3.0e38f;
    int   bpos = 0;

    for (int jt = 0; jt < BN; jt += 256) {
        const int rowc = (b * BN + jt + threadIdx.x) * 66;
        spos[threadIdx.x][0] = x[rowc + 0];
        spos[threadIdx.x][1] = x[rowc + 1];
        __syncthreads();
        for (int jj = 0; jj < 256; ++jj) {
            const int j  = jt + jj;
            const float dx = qx - spos[jj][0];
            const float dy = qy - spos[jj][1];
            const float d2 = dx * dx + dy * dy;
            if (d2 < bmax && j != i) {
                // replace current worst (bpos is dynamic: unrolled selects keep
                // bd/bi in VGPRs instead of spilling to scratch)
#pragma unroll
                for (int t = 0; t < KNN; ++t)
                    if (t == bpos) { bd[t] = d2; bi[t] = j; }
                bmax = bd[0]; bpos = 0;
#pragma unroll
                for (int t = 1; t < KNN; ++t)
                    if (bd[t] > bmax) { bmax = bd[t]; bpos = t; }
            }
        }
        __syncthreads();
    }
    int* kr = knn + (b * BN + i) * KNN;
#pragma unroll
    for (int t = 0; t < KNN; ++t) kr[t] = bi[t];
}

// -------------------------------------------------------------------------
// Kernel 2: gather+mean of neighbor features, then 64->32 (ReLU) -> 64 MLP
// via V_WMMA_F32_16X16X32_F16. One wave handles 16 points (M=16 tile).
// -------------------------------------------------------------------------

// Fuse two 16B LDS loads into one 16-half fragment.
__device__ __forceinline__ v16h ld_frag(const _Float16* p0, const _Float16* p1) {
    v8h lo = *(const v8h*)p0;
    v8h hi = *(const v8h*)p1;
    return __builtin_shufflevector(lo, hi, 0, 1, 2, 3, 4, 5, 6, 7,
                                           8, 9, 10, 11, 12, 13, 14, 15);
}

__global__ __launch_bounds__(128) void gravnet_mlp_kernel(
    const float* __restrict__ x, const int* __restrict__ knn,
    const float* __restrict__ W1, const float* __restrict__ b1,
    const float* __restrict__ W2, const float* __restrict__ b2,
    float* __restrict__ out) {
    // weights stored transposed so B fragments are contiguous in K
    __shared__ alignas(16) _Float16 w1t[HID][NFEAT];   // w1t[n][k] = W1[k][n]
    __shared__ alignas(16) _Float16 w2t[NFEAT][HID];   // w2t[n][k] = W2[k][n]
    __shared__ alignas(16) _Float16 aggh[4][16][NFEAT];
    __shared__ alignas(16) _Float16 hh[4][16][HID];

    const int tid  = threadIdx.x;
    const int lane = tid & 31;
    const int wid  = tid >> 5;

    for (int idx = tid; idx < HID * NFEAT; idx += 128) {
        const int n = idx >> 6, k = idx & 63;
        w1t[n][k] = (_Float16)W1[k * HID + n];
    }
    for (int idx = tid; idx < NFEAT * HID; idx += 128) {
        const int n = idx >> 5, k = idx & 31;
        w2t[n][k] = (_Float16)W2[k * NFEAT + n];
    }
    __syncthreads();

    const int p0    = blockIdx.x * 64 + wid * 16;  // global point base of this wave
    const int b     = p0 >> 11;                    // 2048 points per batch
    const int nbase = p0 & (BN - 1);

    // ---- neighbor-mean aggregation: each lane owns 2 feature columns ----
    const int c0 = lane * 2;
    for (int m = 0; m < 16; ++m) {
        const int  n  = nbase + m;
        const int* kr = knn + (b * BN + n) * KNN;
        float ax = 0.f, ay = 0.f;
#pragma unroll
        for (int t = 0; t < KNN; ++t) {
            const int nbr = kr[t];                       // wave-uniform load
            const float* fr = x + (b * BN + nbr) * 66 + 2 + c0;
            ax += fr[0];
            ay += fr[1];
        }
        v2h pk = { (_Float16)(ax * 0.0625f), (_Float16)(ay * 0.0625f) };
        *(v2h*)&aggh[wid][m][c0] = pk;
    }
    __syncthreads();

    // ---- WMMA A fragments: 16-bit A 16x32 layout (ISA 7.12.2) ----
    // lane<16:  e0..7 -> K=0..7,  e8..15 -> K=16..23  (offset 0 / 16)
    // lane>=16: e0..7 -> K=8..15, e8..15 -> K=24..31  (offset 8 / 24)
    const int mrow  = lane & 15;
    const int koffA = (lane < 16) ? 0 : 8;
    const int koffB = (lane < 16) ? 0 : 8;    // B 32x16: same K order as A (offset 0/16 or 8/24)
    const int mc    = (lane < 16) ? 0 : 8;    // C/D row base

    const _Float16* arow = &aggh[wid][mrow][0];
    const v16h a0 = ld_frag(arow + koffA,      arow + koffA + 16);       // K=0..31
    const v16h a1 = ld_frag(arow + 32 + koffA, arow + 32 + koffA + 16);  // K=32..63

    // ---- GEMM1: agg(16x64) @ W1(64x32) + b1, ReLU ----
    v8f acc[2];
#pragma unroll
    for (int nt = 0; nt < 2; ++nt) {
        const _Float16* brow = &w1t[nt * 16 + mrow][0];
        const v16h bk0 = ld_frag(brow + koffB,      brow + koffB + 16);       // K=0..31
        const v16h bk1 = ld_frag(brow + 32 + koffB, brow + 32 + koffB + 16);  // K=32..63
        const float bias = b1[nt * 16 + mrow];
        v8f c;
#pragma unroll
        for (int r = 0; r < 8; ++r) c[r] = bias;
        c = __builtin_amdgcn_wmma_f32_16x16x32_f16(false, a0, false, bk0,
                                                   (short)0, c, false, false);
        c = __builtin_amdgcn_wmma_f32_16x16x32_f16(false, a1, false, bk1,
                                                   (short)0, c, false, false);
        acc[nt] = c;
    }

    // ReLU, convert to f16, transpose C-layout -> A-layout through LDS
#pragma unroll
    for (int nt = 0; nt < 2; ++nt)
#pragma unroll
        for (int r = 0; r < 8; ++r) {
            float v = acc[nt][r];
            v = v > 0.f ? v : 0.f;
            hh[wid][mc + r][nt * 16 + mrow] = (_Float16)v;
        }
    __syncthreads();

    // ---- GEMM2: h(16x32) @ W2(32x64) + b2 ----
    const _Float16* hrow = &hh[wid][mrow][0];
    const v16h ah = ld_frag(hrow + koffA, hrow + koffA + 16);

    v8f o[4];
#pragma unroll
    for (int nt = 0; nt < 4; ++nt) {
        const _Float16* brow = &w2t[nt * 16 + mrow][0];
        const v16h bf = ld_frag(brow + koffB, brow + koffB + 16);
        const float bias = b2[nt * 16 + mrow];
        v8f c;
#pragma unroll
        for (int r = 0; r < 8; ++r) c[r] = bias;
        c = __builtin_amdgcn_wmma_f32_16x16x32_f16(false, ah, false, bf,
                                                   (short)0, c, false, false);
        o[nt] = c;
    }

    // ---- write output: pos passthrough + updated features ----
    {
        const int m = lane >> 1, comp = lane & 1;   // 32 lanes cover 16 rows x 2
        const int rb = (b * BN + nbase + m) * 66;
        out[rb + comp] = x[rb + comp];
    }
#pragma unroll
    for (int nt = 0; nt < 4; ++nt)
#pragma unroll
        for (int r = 0; r < 8; ++r) {
            const int rb = (b * BN + nbase + mc + r) * 66;
            out[rb + 2 + nt * 16 + mrow] = o[nt][r];
        }
}

extern "C" void kernel_launch(void* const* d_in, const int* in_sizes, int n_in,
                              void* d_out, int out_size, void* d_ws, size_t ws_size,
                              hipStream_t stream) {
    (void)in_sizes; (void)n_in; (void)out_size; (void)ws_size;
    const float* x  = (const float*)d_in[0];
    const float* W1 = (const float*)d_in[1];
    const float* b1 = (const float*)d_in[2];
    const float* W2 = (const float*)d_in[3];
    const float* b2 = (const float*)d_in[4];
    float* out = (float*)d_out;
    int*   knn = (int*)d_ws;   // B*N*K ints = 1 MB scratch

    knn_kernel<<<dim3(BATCH * (BN / 256)), dim3(256), 0, stream>>>(x, knn);
    gravnet_mlp_kernel<<<dim3((BATCH * BN) / 64), dim3(128), 0, stream>>>(
        x, knn, W1, b1, W2, b2, out);
}
